// LSTM_1881195676405
// MI455X (gfx1250) — matrix operand order repaired and checked by reading, not compiled, not run on hardware
//
#include <hip/hip_runtime.h>

// ---------------------------------------------------------------------------
// Problem constants
// ---------------------------------------------------------------------------
#define B_   256
#define D_   64
#define T_   336
#define H_   512
#define LP_  96
#define BT_  16      // batch rows per workgroup
#define G4H  2048    // 4*H
#define K0_  576     // D + H   (layer0 concat K)
#define K1_  1024    // H + H   (layer1 concat K)
#define SXS  80      // LDS row stride (elems) for x tile   (16B aligned, padded)
#define SHS  520     // LDS row stride (elems) for h tiles  (16B aligned, padded)

typedef __attribute__((ext_vector_type(8)))  float  v8f;
typedef __attribute__((ext_vector_type(16))) __bf16 v16bf;
typedef __attribute__((ext_vector_type(8)))  __bf16 v8bf;

union ABf { v16bf v; v8bf h[2]; };

__device__ __forceinline__ __bf16 f2bf(float x) { return (__bf16)x; }
__device__ __forceinline__ v8f splat8(float x) { v8f r = {x,x,x,x,x,x,x,x}; return r; }

// Fast activations: hardware rcp (v_rcp_f32) instead of IEEE divide, hardware
// v_tanh_f32 on gfx1250.
__device__ __forceinline__ float fast_rcp(float x) { return __builtin_amdgcn_rcpf(x); }
__device__ __forceinline__ float sigm(float x) { return fast_rcp(1.f + __expf(-x)); }
__device__ __forceinline__ float tanh_fast(float x) {
#if __has_builtin(__builtin_amdgcn_tanhf)
  return __builtin_amdgcn_tanhf(x);
#elif __has_builtin(__builtin_amdgcn_tanh_f32)
  return __builtin_amdgcn_tanh_f32(x);
#else
  return 1.f - 2.f * fast_rcp(1.f + __expf(2.f * x));
#endif
}

__device__ __forceinline__ v8f wmma_bf16(const ABf& a, const ABf& b, v8f c) {
  // D = A(16x32 bf16) * B(32x16 bf16) + C(16x16 f32)
  return __builtin_amdgcn_wmma_f32_16x16x32_bf16(
      /*neg_a=*/false, a.v, /*neg_b=*/false, b.v,
      /*c_mod=*/(short)0, c, /*reuse_a=*/false, /*reuse_b=*/false);
}

// A-fragment from LDS (16-bit 16x32 A layout: lane=M, chunks K[+lhi*8..+8) and +16)
#define LOAD_A(frag, base, stride, kb)                                  \
  do {                                                                  \
    const __bf16* _p = (base) + (size_t)l15 * (stride) + (kb) + lhi * 8;\
    (frag).h[0] = *(const v8bf*)(_p);                                   \
    (frag).h[1] = *(const v8bf*)(_p + 16);                              \
  } while (0)

// B-fragment from global bf16 weights, row-major [n][k] (32x16 B layout:
// lane=N, chunks K[+lhi*8..+8) and +16, matching the A layout)
#define LOAD_B(frag, Wp, Kc, nrow, kb)                                  \
  do {                                                                  \
    const __bf16* _q = (Wp) + (size_t)(nrow) * (Kc) + (kb) + lhi * 8;   \
    (frag).h[0] = *(const v8bf*)(_q);                                   \
    (frag).h[1] = *(const v8bf*)(_q + 16);                              \
  } while (0)

// A-region selectors for the two concatenated-K GEMMs
#define LOAD_A_L0(frag, kt)                                             \
  do { if ((kt) < 2) { LOAD_A(frag, sx,  SXS, 32 * (kt)); }             \
       else          { LOAD_A(frag, sh0, SHS, 32 * ((kt) - 2)); } } while (0)
#define LOAD_A_L1(frag, kt)                                             \
  do { if ((kt) < 16) { LOAD_A(frag, sh0, SHS, 32 * (kt)); }            \
       else           { LOAD_A(frag, sh1, SHS, 32 * ((kt) - 16)); } } while (0)

// Software-pipelined GEMM over ktiles. B fragments live in a 4-slot ring with
// purely compile-time slot indices: WMMA for gate g consumes slot g while the
// load for fragment two-ahead is issued into slot (g+2)&3. Prefetch distance =
// 2 fragments (4 outstanding b128 loads), no conditionally-written slots, no
// ring pointer -> SROA keeps everything in registers. Tail prefetches clamp to
// the last ktile (redundant but valid loads).
#define GEMM_PIPE(NKT, LOADA, W, Kc, biasv)                             \
  do {                                                                  \
    _Pragma("unroll")                                                   \
    for (int g = 0; g < 4; ++g) acc[g] = splat8((biasv)[g]);            \
    ABf bX[4];                                                          \
    LOAD_B(bX[0], W, Kc, 0 * H_ + jbase + l15, 0);                      \
    LOAD_B(bX[1], W, Kc, 1 * H_ + jbase + l15, 0);                      \
    _Pragma("unroll 2")                                                 \
    for (int kt = 0; kt < (NKT); ++kt) {                                \
      ABf a;                                                            \
      LOADA(a, kt);                                                     \
      _Pragma("unroll")                                                 \
      for (int g = 0; g < 4; ++g) {                                     \
        int ng  = (g + 2) & 3;                                          \
        int nkt = kt + ((g >= 2) ? 1 : 0);                              \
        if (nkt >= (NKT)) nkt = (NKT) - 1;                              \
        LOAD_B(bX[ng], W, Kc, ng * H_ + jbase + l15, 32 * nkt);         \
        acc[g] = wmma_bf16(a, bX[g], acc[g]);                           \
      }                                                                 \
    }                                                                   \
  } while (0)

// ---------------------------------------------------------------------------
// Weight / bias packing (fp32 -> bf16, concat [W_ih | W_hh] row-wise)
// ---------------------------------------------------------------------------
__global__ void pack_w0(const float* __restrict__ Wih0, const float* __restrict__ Whh0,
                        __bf16* __restrict__ W0c) {
  int idx = blockIdx.x * 256 + threadIdx.x;
  if (idx >= G4H * K0_) return;
  int n = idx / K0_, k = idx - n * K0_;
  float v = (k < D_) ? Wih0[n * D_ + k] : Whh0[n * H_ + (k - D_)];
  W0c[idx] = f2bf(v);
}

__global__ void pack_w1(const float* __restrict__ Wih1, const float* __restrict__ Whh1,
                        __bf16* __restrict__ W1c) {
  int idx = blockIdx.x * 256 + threadIdx.x;
  if (idx >= G4H * K1_) return;
  int n = idx >> 10, k = idx & 1023;
  float v = (k < H_) ? Wih1[n * H_ + k] : Whh1[n * H_ + (k - H_)];
  W1c[idx] = f2bf(v);
}

__global__ void pack_wo(const float* __restrict__ Wout, __bf16* __restrict__ Wo) {
  int idx = blockIdx.x * 256 + threadIdx.x;
  if (idx >= D_ * H_) return;
  Wo[idx] = f2bf(Wout[idx]);
}

__global__ void pack_bias(const float* __restrict__ bi0, const float* __restrict__ bh0,
                          const float* __restrict__ bi1, const float* __restrict__ bh1,
                          float* __restrict__ bias0, float* __restrict__ bias1) {
  int i = blockIdx.x * 256 + threadIdx.x;
  if (i >= G4H) return;
  bias0[i] = bi0[i] + bh0[i];
  bias1[i] = bi1[i] + bh1[i];
}

// ---------------------------------------------------------------------------
// Persistent LSTM kernel: 16 workgroups (one per 16-row batch tile),
// 1024 threads = 32 wave32s; wave w owns hidden units [16w, 16w+16).
// ---------------------------------------------------------------------------
__global__ __launch_bounds__(1024, 1) void lstm_persist(
    const float* __restrict__ x,      // (B, D, T) fp32
    const float* __restrict__ bout,   // (D) fp32
    const __bf16* __restrict__ W0c,   // (4H, D+H) bf16
    const __bf16* __restrict__ W1c,   // (4H, 2H)  bf16
    const __bf16* __restrict__ Wo,    // (D, H)    bf16
    const float* __restrict__ bias0,  // (4H)
    const float* __restrict__ bias1,  // (4H)
    float* __restrict__ outGF,        // (B, H)
    float* __restrict__ outPred)      // (B, D, LP)
{
  __shared__ __align__(16) __bf16 sx[BT_ * SXS];
  __shared__ __align__(16) __bf16 sh0[BT_ * SHS];
  __shared__ __align__(16) __bf16 sh1[BT_ * SHS];

  const int tid   = threadIdx.x;
  const int wave  = tid >> 5;
  const int lane  = tid & 31;
  const int l15   = lane & 15;
  const int lhi   = lane >> 4;        // 0 or 1
  const int bbase = blockIdx.x * BT_;
  const int jbase = wave * 16;        // hidden-unit base for this wave

  // zero LDS state (h0 = h1 = 0)
  for (int i = tid; i < BT_ * SXS; i += 1024) sx[i] = f2bf(0.f);
  for (int i = tid; i < BT_ * SHS; i += 1024) { sh0[i] = f2bf(0.f); sh1[i] = f2bf(0.f); }

  // per-lane gate biases (column n = gate*H + jbase + l15), splatted over rows
  float b0v[4], b1v[4];
#pragma unroll
  for (int g = 0; g < 4; ++g) {
    b0v[g] = bias0[g * H_ + jbase + l15];
    b1v[g] = bias1[g * H_ + jbase + l15];
  }
  float bov = (wave < 4) ? bout[wave * 16 + l15] : 0.f;

  // cell states in registers, C-matrix layout: element r -> (m = lhi*8+r, n = jbase+l15)
  float c0r[8], c1r[8];
#pragma unroll
  for (int r = 0; r < 8; ++r) { c0r[r] = 0.f; c1r[r] = 0.f; }

  for (int step = 0; step < T_ + LP_; ++step) {
    // ---- stage the layer-0 input tile ----
    if (step < T_) {
      int m = tid >> 6, d = tid & 63;   // 1024 threads -> 16x64 tile
      float xv = x[((size_t)(bbase + m) * D_ + d) * T_ + step];
      sx[m * SXS + d] = f2bf(xv);
    } else if (step == T_) {
      int m = tid >> 6, d = tid & 63;   // decoder starts from zeros
      sx[m * SXS + d] = f2bf(0.f);
    }
    __syncthreads();  // sx / sh0 / sh1 stable

    // ================= layer 0: g = [x|h0] @ W0c^T + b0 =================
    v8f acc[4];
    GEMM_PIPE(18, LOAD_A_L0, W0c, K0_, b0v);   // K = 64 + 512 = 18 * 32
    __syncthreads();  // all reads of sh0 done before rewrite

    // elementwise LSTM cell, layer 0
#pragma unroll
    for (int r = 0; r < 8; ++r) {
      float iv = sigm(acc[0][r]);
      float fv = sigm(acc[1][r]);
      float gv = tanh_fast(acc[2][r]);
      float ov = sigm(acc[3][r]);
      float cn = fv * c0r[r] + iv * gv;
      float hn = ov * tanh_fast(cn);
      c0r[r] = cn;
      int m = lhi * 8 + r;
      sh0[m * SHS + jbase + l15] = f2bf(hn);
    }
    __syncthreads();  // sh0 ready

    // ================= layer 1: g = [h0|h1] @ W1c^T + b1 =================
    GEMM_PIPE(32, LOAD_A_L1, W1c, K1_, b1v);   // K = 512 + 512 = 32 * 32
    __syncthreads();  // all reads of sh1 done before rewrite

    // elementwise LSTM cell, layer 1 (+ global_feat at last encoder step)
#pragma unroll
    for (int r = 0; r < 8; ++r) {
      float iv = sigm(acc[0][r]);
      float fv = sigm(acc[1][r]);
      float gv = tanh_fast(acc[2][r]);
      float ov = sigm(acc[3][r]);
      float cn = fv * c1r[r] + iv * gv;
      float hn = ov * tanh_fast(cn);
      c1r[r] = cn;
      int m = lhi * 8 + r;
      sh1[m * SHS + jbase + l15] = f2bf(hn);
      if (step == T_ - 1)
        outGF[(size_t)(bbase + m) * H_ + jbase + l15] = hn;
    }

    // ================= decoder head: pred = h1 @ Wout^T + b_out ==========
    if (step >= T_) {
      __syncthreads();  // sh1 ready for the head GEMM
      if (wave < 4) {   // N = 64 -> 4 tiles, one per wave (uniform per-wave branch)
        v8f pacc = splat8(bov);
        ABf hb[2];
        LOAD_B(hb[0], Wo, H_, wave * 16 + l15, 0);
#pragma unroll 2
        for (int kt = 0; kt < 16; ++kt) {    // K = 512
          ABf a;
          LOAD_A(a, sh1, SHS, 32 * kt);
          int nkt = (kt + 1 < 16) ? kt + 1 : 15;
          LOAD_B(hb[(kt + 1) & 1], Wo, H_, wave * 16 + l15, 32 * nkt);
          pacc = wmma_bf16(a, hb[kt & 1], pacc);
        }
        int s = step - T_;
        int dcol = wave * 16 + l15;
#pragma unroll
        for (int r = 0; r < 8; ++r) {
          int m = lhi * 8 + r;
          float pv = pacc[r];
          outPred[(size_t)(bbase + m) * (D_ * LP_) + (size_t)dcol * LP_ + s] = pv;
          sx[m * SXS + dcol] = f2bf(pv);   // feed next decoder step
        }
      }
    }
    // top-of-loop __syncthreads orders sx/sh writes vs. next step's reads
  }
}

// ---------------------------------------------------------------------------
// Host-side launch
// ---------------------------------------------------------------------------
extern "C" void kernel_launch(void* const* d_in, const int* in_sizes, int n_in,
                              void* d_out, int out_size, void* d_ws, size_t ws_size,
                              hipStream_t stream) {
  (void)in_sizes; (void)n_in; (void)out_size; (void)ws_size;

  const float* x    = (const float*)d_in[0];
  const float* Wih0 = (const float*)d_in[1];
  const float* Whh0 = (const float*)d_in[2];
  const float* bih0 = (const float*)d_in[3];
  const float* bhh0 = (const float*)d_in[4];
  const float* Wih1 = (const float*)d_in[5];
  const float* Whh1 = (const float*)d_in[6];
  const float* bih1 = (const float*)d_in[7];
  const float* bhh1 = (const float*)d_in[8];
  const float* Wout = (const float*)d_in[9];
  const float* bout = (const float*)d_in[10];

  // workspace layout (bytes): bf16 weights + fused biases (~6.6 MB)
  char* ws = (char*)d_ws;
  __bf16* W0c   = (__bf16*)(ws + 0);              // 2048*576*2  = 2,359,296
  __bf16* W1c   = (__bf16*)(ws + 2359296);        // 2048*1024*2 = 4,194,304
  __bf16* Wo    = (__bf16*)(ws + 6553600);        // 64*512*2    =    65,536
  float*  bias0 = (float*)(ws + 6619136);         // 2048*4
  float*  bias1 = (float*)(ws + 6627328);         // 2048*4

  pack_w0<<<(G4H * K0_ + 255) / 256, 256, 0, stream>>>(Wih0, Whh0, W0c);
  pack_w1<<<(G4H * K1_ + 255) / 256, 256, 0, stream>>>(Wih1, Whh1, W1c);
  pack_wo<<<(D_ * H_ + 255) / 256, 256, 0, stream>>>(Wout, Wo);
  pack_bias<<<(G4H + 255) / 256, 256, 0, stream>>>(bih0, bhh0, bih1, bhh1, bias0, bias1);

  float* outGF = (float*)d_out;              // (B, H)
  float* outPr = outGF + (size_t)B_ * H_;    // (B, D, LP)
  lstm_persist<<<B_ / BT_, 1024, 0, stream>>>(x, bout, W0c, W1c, Wo, bias0, bias1,
                                              outGF, outPr);
}
